// InputProjection_41120016892313
// MI455X (gfx1250) — compile-verified, matrix operand order checked by reading, not run on hardware
//
#include <hip/hip_runtime.h>
#include <hip/hip_bf16.h>

typedef __attribute__((ext_vector_type(16))) _Float16     v16h;
typedef __attribute__((ext_vector_type(8)))  float        v8f;
typedef __attribute__((ext_vector_type(4)))  unsigned int v4u;

#define D_IN        25
#define D_PROJ      128
#define ROWS_TOTAL  (256 * 2048)
#define WAVES       4
#define BLOCK       (WAVES * 32)
#define ROWS_PER_BLOCK (WAVES * 16)

// w1 fragments: 8 tiles * 32 lanes * 16 f16 = 4096 f16
// w2 fragments: 32 tiles * 32 lanes * 16 f16 = 16384 f16
#define W1_FRAG_ELEMS 4096
#define W2_FRAG_ELEMS 16384

union V16U { v16h h; v4u u[2]; };

// Branchless GELU (tanh form) on hardware transcendentals:
//   z  = sqrt(2/pi) * (x + 0.044715 x^3)
//   q  = exp2(-2*log2(e)*|z|)            (v_exp_f32 is base-2)
//   th = sign(z) * (1-q) * rcp(1+q)      (v_rcp_f32)
// Max |error| vs exact erf-GELU ~3e-4, below the f16-matmul noise floor.
__device__ __forceinline__ float gelu_fast(float x) {
  float x2 = x * x;
  float z  = 0.7978845608028654f * __builtin_fmaf(0.044715f * x2, x, x);
  float az = __builtin_fabsf(z);
  float q  = __builtin_amdgcn_exp2f(az * -2.8853900817779268f); // -2/ln2
  float th = (1.0f - q) * __builtin_amdgcn_rcpf(1.0f + q);
  th = __builtin_copysignf(th, z);
  return 0.5f * x * (1.0f + th);
}

// ---------------------------------------------------------------------------
// Prep: convert fp32 weights into f16 WMMA B-fragment order in workspace.
// B frag layout (16x16x32 f16): lane (half*16 + c) holds column c of the tile;
// elem i<8 -> K = kb+i, elem i>=8 -> K = kb+16+(i-8), kb = half ? 8 : 0.
// w1 K-dim padded 25 -> 32 with zeros.
// ---------------------------------------------------------------------------
__global__ void prep_weights_kernel(const float* __restrict__ w1,
                                    const float* __restrict__ w2,
                                    _Float16* __restrict__ ws) {
  int e = blockIdx.x * blockDim.x + threadIdx.x;
  if (e >= W1_FRAG_ELEMS + W2_FRAG_ELEMS) return;
  if (e < W1_FRAG_ELEMS) {
    int t = e >> 9, rem = e & 511, lane = rem >> 4, i = rem & 15;
    int c = lane & 15, kb = (lane < 16) ? 0 : 8;
    int k = (i < 8) ? (kb + i) : (kb + 16 + (i - 8));
    float v = (k < D_IN) ? w1[k * D_PROJ + t * 16 + c] : 0.0f;
    ws[e] = (_Float16)v;
  } else {
    int e2 = e - W1_FRAG_ELEMS;
    int tile = e2 >> 9, rem = e2 & 511, lane = rem >> 4, i = rem & 15;
    int kc = tile >> 3, t = tile & 7;
    int c = lane & 15, kb = (lane < 16) ? 0 : 8;
    int k = kc * 32 + ((i < 8) ? (kb + i) : (kb + 16 + (i - 8)));
    ws[e] = (_Float16)w2[k * D_PROJ + t * 16 + c];
  }
}

// ---------------------------------------------------------------------------
// Fused LayerNorm -> (xn @ w1 + b1) -> GELU -> (h @ w2 + b2)
// 4 waves/block, 16 rows/wave, all matmuls via v_wmma_f32_16x16x32_f16.
// ---------------------------------------------------------------------------
__global__ __launch_bounds__(BLOCK)
void fused_ln_mlp_kernel(const float* __restrict__ x,
                         const float* __restrict__ gamma,
                         const float* __restrict__ beta,
                         const _Float16* __restrict__ wfrag,
                         const float* __restrict__ b1,
                         const float* __restrict__ b2,
                         float* __restrict__ out) {
  __shared__ __align__(16) _Float16 sW1[W1_FRAG_ELEMS];          //  8 KB
  __shared__ __align__(16) _Float16 sW2[W2_FRAG_ELEMS];          // 32 KB
  __shared__ __align__(16) _Float16 sXn[WAVES * 16 * 32];        //  4 KB
  __shared__ __align__(16) _Float16 sH[WAVES * 16 * D_PROJ];     // 16 KB
  __shared__ float sB1[D_PROJ];
  __shared__ float sB2[D_PROJ];

  const int tid  = threadIdx.x;
  const int wave = tid >> 5;
  const int lane = tid & 31;

  // Cooperative weight-fragment load, 128-bit granularity.
  {
    const v4u* src = (const v4u*)wfrag;
    v4u* dW1 = (v4u*)sW1;
    v4u* dW2 = (v4u*)sW2;
    #pragma unroll
    for (int i = tid; i < W1_FRAG_ELEMS / 8; i += BLOCK) dW1[i] = src[i];
    #pragma unroll
    for (int i = tid; i < W2_FRAG_ELEMS / 8; i += BLOCK)
      dW2[i] = src[W1_FRAG_ELEMS / 8 + i];
    sB1[tid] = b1[tid];
    sB2[tid] = b2[tid];
  }
  __syncthreads();

  const long rowBase = (long)blockIdx.x * ROWS_PER_BLOCK + (long)wave * 16;

  // ---- LayerNorm: lanes 0..15 each own one row (25 elems, eps = 1e-5) ----
  if (lane < 16) {
    const float* xr = x + (rowBase + lane) * D_IN;
    float xv[D_IN];
    float s = 0.0f;
    #pragma unroll
    for (int k = 0; k < D_IN; ++k) { xv[k] = xr[k]; s += xv[k]; }
    const float mu = s * (1.0f / D_IN);
    float var = 0.0f;
    #pragma unroll
    for (int k = 0; k < D_IN; ++k) { float d = xv[k] - mu; var += d * d; }
    const float rstd = rsqrtf(var * (1.0f / D_IN) + 1e-5f);
    _Float16* xnr = &sXn[(wave * 16 + lane) * 32];
    #pragma unroll
    for (int k = 0; k < D_IN; ++k)
      xnr[k] = (_Float16)((xv[k] - mu) * rstd * gamma[k] + beta[k]);
    #pragma unroll
    for (int k = D_IN; k < 32; ++k) xnr[k] = (_Float16)0.0f;
  }
  // sXn produced and consumed within the same wave: DS counter ordering only.

  const int r    = lane & 15;              // row within tile
  const int kb   = (lane < 16) ? 0 : 8;    // K sub-block for A/B frags
  const int c16  = lane & 15;              // column within N-tile (C layout)
  const int mb   = (lane < 16) ? 0 : 8;    // row base in C layout

  // ---- GEMM1: (16x32) x (32x128), K padded 25->32, one WMMA per N-tile ----
  V16U a1;
  {
    const _Float16* ap = &sXn[(wave * 16 + r) * 32];
    a1.u[0] = *(const v4u*)(ap + kb);
    a1.u[1] = *(const v4u*)(ap + kb + 16);
  }

  v8f acc[8];
  #pragma unroll
  for (int t = 0; t < 8; ++t) {
    V16U b;
    const v4u* bp = (const v4u*)&sW1[(t * 32 + lane) * 16];
    b.u[0] = bp[0];
    b.u[1] = bp[1];
    v8f czero = {};
    acc[t] = __builtin_amdgcn_wmma_f32_16x16x32_f16(
        false, a1.h, false, b.h, (short)0, czero, false, false);
  }

  // ---- bias + fast GELU, restage h into LDS row-major [16][128] ----
  _Float16* hw = &sH[wave * 16 * D_PROJ];
  #pragma unroll
  for (int t = 0; t < 8; ++t) {
    const int col = t * 16 + c16;
    const float bias = sB1[col];
    #pragma unroll
    for (int j = 0; j < 8; ++j) {
      float h = acc[t][j] + bias;
      hw[(mb + j) * D_PROJ + col] = (_Float16)gelu_fast(h);
    }
  }

  // ---- GEMM2: (16x128) x (128x128), 4 K-chunks x 8 N-tiles ----
  v8f acc2[8];
  #pragma unroll
  for (int t = 0; t < 8; ++t) { v8f z = {}; acc2[t] = z; }

  #pragma unroll
  for (int kc = 0; kc < 4; ++kc) {
    V16U a2;
    const _Float16* ap = &hw[r * D_PROJ + kc * 32];
    a2.u[0] = *(const v4u*)(ap + kb);
    a2.u[1] = *(const v4u*)(ap + kb + 16);
    #pragma unroll
    for (int t = 0; t < 8; ++t) {
      V16U b;
      const v4u* bp = (const v4u*)&sW2[((kc * 8 + t) * 32 + lane) * 16];
      b.u[0] = bp[0];
      b.u[1] = bp[1];
      acc2[t] = __builtin_amdgcn_wmma_f32_16x16x32_f16(
          false, a2.h, false, b.h, (short)0, acc2[t], false, false);
    }
  }

  // ---- epilogue: + b2, fp32 store ----
  #pragma unroll
  for (int t = 0; t < 8; ++t) {
    const int col = t * 16 + c16;
    const float bias = sB2[col];
    float* orow = out + (rowBase + mb) * D_PROJ + col;
    #pragma unroll
    for (int j = 0; j < 8; ++j)
      orow[(long)j * D_PROJ] = acc2[t][j] + bias;
  }
}

// ---------------------------------------------------------------------------
extern "C" void kernel_launch(void* const* d_in, const int* in_sizes, int n_in,
                              void* d_out, int out_size, void* d_ws, size_t ws_size,
                              hipStream_t stream) {
  const float* x     = (const float*)d_in[0];
  const float* gamma = (const float*)d_in[1];
  const float* beta  = (const float*)d_in[2];
  const float* w1    = (const float*)d_in[3];
  const float* b1    = (const float*)d_in[4];
  const float* w2    = (const float*)d_in[5];
  const float* b2    = (const float*)d_in[6];
  float*       out   = (float*)d_out;
  _Float16*    wfrag = (_Float16*)d_ws;   // 40 KB of f16 weight fragments

  const int prep_elems = W1_FRAG_ELEMS + W2_FRAG_ELEMS;
  prep_weights_kernel<<<(prep_elems + 255) / 256, 256, 0, stream>>>(w1, w2, wfrag);

  const int grid = ROWS_TOTAL / ROWS_PER_BLOCK;  // 524288 / 64 = 8192
  fused_ln_mlp_kernel<<<grid, BLOCK, 0, stream>>>(x, gamma, beta, wfrag,
                                                  b1, b2, out);
}
